// MixedHeads_21663815041813
// MI455X (gfx1250) — matrix operand order repaired and checked by reading, not compiled, not run on hardware
//
#include <hip/hip_runtime.h>

#define EMBED 1024
#define NHEAD 16
#define HSIZE 64
#define MAXE  2048
#define MAXHS 128
#define BB    4
#define TT    2048

typedef _Float16 v8h  __attribute__((ext_vector_type(8)));
typedef _Float16 v16h __attribute__((ext_vector_type(16)));
typedef float    v8f  __attribute__((ext_vector_type(8)));

static __device__ __forceinline__ v8f wmma_f16(v16h a, v16h b, v8f c) {
  // D = A(16x32 f16) * B(32x16 f16) + C(16x16 f32)
  return __builtin_amdgcn_wmma_f32_16x16x32_f16(false, a, false, b, (short)0, c,
                                                false, false);
}

// ---------------------------------------------------------------------------
// Kernel 0a: convert x[:, :, :1024] f32 -> f16 once (bandwidth-bound).
// ---------------------------------------------------------------------------
__global__ void cvt_x_kernel(const float* __restrict__ x,
                             _Float16* __restrict__ xh) {
  const int i = blockIdx.x * blockDim.x + threadIdx.x;  // 2M quads
  const int total4 = BB * TT * EMBED / 4;
  if (i >= total4) return;
  const int e  = (i & (EMBED / 4 - 1)) * 4;  // EMBED/4 = 256
  const int bt = i >> 8;                     // b*TT + t
  const float4 v = *(const float4*)(x + (size_t)bt * MAXE + e);
  _Float16* d = xh + (size_t)bt * EMBED + e;
  d[0] = (_Float16)v.x; d[1] = (_Float16)v.y;
  d[2] = (_Float16)v.z; d[3] = (_Float16)v.w;
}

// ---------------------------------------------------------------------------
// Kernel 0b: convert W{q,k,v}[:, :64, :1024] f32 -> f16 packed [3][16][64][1024].
// ---------------------------------------------------------------------------
__global__ void cvt_w_kernel(const float* __restrict__ Wq,
                             const float* __restrict__ Wk,
                             const float* __restrict__ Wv,
                             _Float16* __restrict__ wh) {
  const int i = blockIdx.x * blockDim.x + threadIdx.x;  // 786432 quads
  const int total4 = 3 * NHEAD * HSIZE * EMBED / 4;
  if (i >= total4) return;
  const int e   = (i & 255) * 4;       // EMBED/4 = 256
  int rem       = i >> 8;              // p*1024 + h*64 + d
  const int d   = rem & 63;  rem >>= 6;
  const int h   = rem & 15;
  const int p   = rem >> 4;
  const float* W = (p == 0) ? Wq : (p == 1) ? Wk : Wv;
  const float4 v = *(const float4*)(W + ((size_t)h * MAXHS + d) * MAXE + e);
  _Float16* dst = wh + (((size_t)p * NHEAD + h) * HSIZE + d) * EMBED + e;
  dst[0] = (_Float16)v.x; dst[1] = (_Float16)v.y;
  dst[2] = (_Float16)v.z; dst[3] = (_Float16)v.w;
}

// ---------------------------------------------------------------------------
// Kernel 1: QKV projection, all-f16 operands.  One wave -> 32(t) x 64(d) tile
// of one projection for one (b,h); each W B-fragment feeds two A-tiles.
// q,k stored f16 [b][h][t][64] (q pre-scaled 1/8); v stored transposed
// [b][h][d][T] so attention PV B-fragments are contiguous.
// ---------------------------------------------------------------------------
__global__ __launch_bounds__(128)
void qkv_proj_kernel(const _Float16* __restrict__ xh,
                     const _Float16* __restrict__ wh,
                     _Float16* __restrict__ qh,
                     _Float16* __restrict__ kh,
                     _Float16* __restrict__ vth) {
  const int lane = threadIdx.x & 31;
  const int wave = threadIdx.x >> 5;
  const int ln   = lane & 15;
  const int g    = lane >> 4;

  int w = blockIdx.x * 4 + wave;          // 12288 waves total
  const int trow = w & 63;   w >>= 6;     // T/32 = 64 row tiles
  const int h    = w & 15;   w >>= 4;
  const int proj = w % 3;
  const int b    = w / 3;
  if (b >= BB) return;

  const _Float16* wbase = wh + ((size_t)(proj * NHEAD + h) * HSIZE) * EMBED;
  const _Float16* xr0 = xh + ((size_t)b * TT + (size_t)(trow * 32 + ln)) * EMBED;
  const _Float16* xr1 = xr0 + (size_t)16 * EMBED;

  v8f acc[2][4] = {};                     // [sub-tile st][d0 = 0,16,32,48]
  for (int e0 = 0; e0 < EMBED; e0 += 32) {
    v16h af0, af1;
    {
      v8h a0 = *(const v8h*)(xr0 + e0 + 16 * g);
      v8h a1 = *(const v8h*)(xr0 + e0 + 16 * g + 8);
      v8h b0 = *(const v8h*)(xr1 + e0 + 16 * g);
      v8h b1 = *(const v8h*)(xr1 + e0 + 16 * g + 8);
#pragma unroll
      for (int i = 0; i < 8; ++i) {
        af0[i] = a0[i]; af0[8 + i] = a1[i];
        af1[i] = b0[i]; af1[8 + i] = b1[i];
      }
    }
#pragma unroll
    for (int c = 0; c < 4; ++c) {
      const v16h bf =
          *(const v16h*)(wbase + (size_t)(16 * c + ln) * EMBED + e0 + 16 * g);
      acc[0][c] = wmma_f16(af0, bf, acc[0][c]);
      acc[1][c] = wmma_f16(af1, bf, acc[1][c]);
    }
  }

  const size_t bh = (size_t)b * NHEAD + h;
#pragma unroll
  for (int st = 0; st < 2; ++st) {
    const int tbase = trow * 32 + st * 16;
    if (proj == 2) {
      // v transposed: vth[bh][d][t]; r=0..7 -> contiguous t (16B store)
#pragma unroll
      for (int c = 0; c < 4; ++c) {
        const int d = 16 * c + ln;
        _Float16* dst = vth + (bh * HSIZE + d) * TT + tbase + 8 * g;
        v8h chunk;
#pragma unroll
        for (int r = 0; r < 8; ++r) chunk[r] = (_Float16)acc[st][c][r];
        *(v8h*)dst = chunk;
      }
    } else {
      _Float16* dst = (proj == 0) ? qh : kh;
      const float scale = (proj == 0) ? 0.125f : 1.0f;  // fold 1/sqrt(64)
#pragma unroll
      for (int c = 0; c < 4; ++c)
#pragma unroll
        for (int r = 0; r < 8; ++r) {
          const int t = tbase + r + 8 * g;
          dst[(bh * TT + t) * HSIZE + 16 * c + ln] =
              (_Float16)(acc[st][c][r] * scale);
        }
    }
  }
}

// ---------------------------------------------------------------------------
// Kernel 2: flash attention.  Block = 4 waves = one 64-query tile of one
// (b,h); wave owns 16 query rows.  Key blocks of 32: 4 WMMAs for S, online
// softmax (shfl width 16), S->P A-fragment transpose via 1KB/wave LDS,
// 4 WMMAs for PV.  Next K/V block prefetched (global_prefetch_b8).
// ---------------------------------------------------------------------------
__global__ __launch_bounds__(128)
void attn_kernel(const _Float16* __restrict__ qh,
                 const _Float16* __restrict__ kh,
                 const _Float16* __restrict__ vth,
                 float* __restrict__ out) {
  __shared__ _Float16 plds[4][16][32];

  const int lane = threadIdx.x & 31;
  const int wave = threadIdx.x >> 5;
  const int ln   = lane & 15;
  const int g    = lane >> 4;

  int blk = blockIdx.x;                    // 2048 blocks
  const int qtile = blk & 31;  blk >>= 5;  // T/64 = 32
  const int h     = blk & 15;  blk >>= 4;
  const int b     = blk;
  const size_t bh = (size_t)b * NHEAD + h;
  const int qbase = qtile * 64;
  const int qrow0 = qbase + wave * 16;

  // Q A-fragments (d 0..31 and 32..63), per-lane 16B contiguous chunks
  const _Float16* qrow = qh + (bh * TT + (size_t)(qrow0 + ln)) * HSIZE;
  v16h qa0, qa1;
  {
    v8h a0 = *(const v8h*)(qrow + 16 * g);
    v8h a1 = *(const v8h*)(qrow + 16 * g + 8);
    v8h a2 = *(const v8h*)(qrow + 32 + 16 * g);
    v8h a3 = *(const v8h*)(qrow + 40 + 16 * g);
#pragma unroll
    for (int i = 0; i < 8; ++i) {
      qa0[i] = a0[i]; qa0[8 + i] = a1[i];
      qa1[i] = a2[i]; qa1[8 + i] = a3[i];
    }
  }

  float mi[8], li[8];
  v8f of[4] = {};
#pragma unroll
  for (int r = 0; r < 8; ++r) { mi[r] = -__builtin_inff(); li[r] = 0.0f; }

  const int s_end = qbase + 64;            // same for all 4 waves -> uniform
  for (int s0 = 0; s0 < s_end; s0 += 32) {
    // prefetch next key/value block while this one computes
    if (s0 + 32 < s_end) {
      __builtin_prefetch(kh + (bh * TT + (size_t)(s0 + 32 + lane)) * HSIZE, 0, 3);
      __builtin_prefetch(vth + (bh * HSIZE + lane) * TT + s0 + 32, 0, 3);
      __builtin_prefetch(vth + (bh * HSIZE + 32 + lane) * TT + s0 + 32, 0, 3);
    }

    // ---- S = Q K^T over 32 keys: two N=16 C-fragments ----
    v8f sf[2];
#pragma unroll
    for (int c = 0; c < 2; ++c) {
      const _Float16* krow =
          kh + (bh * TT + (size_t)(s0 + 16 * c + ln)) * HSIZE;
      v16h kb0 = *(const v16h*)(krow + 16 * g);        // d 0..31
      v16h kb1 = *(const v16h*)(krow + 32 + 16 * g);   // d 32..63
      v8f s = {};
      s = wmma_f16(qa0, kb0, s);
      s = wmma_f16(qa1, kb1, s);
      sf[c] = s;
    }

    // ---- causal mask + online softmax ----
#pragma unroll
    for (int r = 0; r < 8; ++r) {
      const int srow = qrow0 + r + 8 * g;
      float sa = sf[0][r], sb = sf[1][r];
      if (s0 + ln > srow)      sa = -__builtin_inff();
      if (s0 + 16 + ln > srow) sb = -__builtin_inff();
      float bm = fmaxf(sa, sb);
#pragma unroll
      for (int m = 1; m < 16; m <<= 1) bm = fmaxf(bm, __shfl_xor(bm, m, 16));
      const float nm = fmaxf(mi[r], bm);
      const float pa = __expf(sa - nm);
      const float pb = __expf(sb - nm);
      float rs = pa + pb;
#pragma unroll
      for (int m = 1; m < 16; m <<= 1) rs += __shfl_xor(rs, m, 16);
      const float alpha = __expf(mi[r] - nm);
      li[r] = li[r] * alpha + rs;
      mi[r] = nm;
#pragma unroll
      for (int c = 0; c < 4; ++c) of[c][r] *= alpha;
      plds[wave][r + 8 * g][ln]      = (_Float16)pa;
      plds[wave][r + 8 * g][16 + ln] = (_Float16)pb;
    }
    __syncthreads();

    // ---- P C-layout -> A-fragment via LDS ----
    v16h pf;
    {
      const _Float16* pr = &plds[wave][ln][0];
      v8h c0 = *(const v8h*)(pr + 16 * g);
      v8h c1 = *(const v8h*)(pr + 16 * g + 8);
#pragma unroll
      for (int i = 0; i < 8; ++i) { pf[i] = c0[i]; pf[8 + i] = c1[i]; }
    }
    __syncthreads();

    // ---- O += P V  (V^T layout -> contiguous 32B B-fragment loads) ----
#pragma unroll
    for (int c = 0; c < 4; ++c) {
      const _Float16* vrow =
          vth + (bh * HSIZE + (size_t)(16 * c + ln)) * TT + s0 + 16 * g;
      v16h vb = *(const v16h*)vrow;
      of[c] = wmma_f16(pf, vb, of[c]);
    }
  }

  // ---- epilogue: normalize and scatter to out[b][t][h*64 + d] ----
#pragma unroll
  for (int r = 0; r < 8; ++r) {
    const int t = qrow0 + r + 8 * g;
    const float inv = 1.0f / li[r];
    float* orow = out + ((size_t)b * TT + t) * MAXE + h * HSIZE;
#pragma unroll
    for (int c = 0; c < 4; ++c) orow[16 * c + ln] = of[c][r] * inv;
  }
}

// ---------------------------------------------------------------------------
// Kernel 3: zero the padded output columns [1024, 2048).
// ---------------------------------------------------------------------------
__global__ void pad_zero_kernel(float* __restrict__ out) {
  const size_t i = (size_t)blockIdx.x * blockDim.x + threadIdx.x;
  const size_t total = (size_t)BB * TT * (MAXE - EMBED);
  if (i < total) {
    const size_t row = i / (MAXE - EMBED);
    const size_t col = i % (MAXE - EMBED);
    out[row * MAXE + EMBED + col] = 0.0f;
  }
}

extern "C" void kernel_launch(void* const* d_in, const int* in_sizes, int n_in,
                              void* d_out, int out_size, void* d_ws,
                              size_t ws_size, hipStream_t stream) {
  const float* x  = (const float*)d_in[0];
  const float* Wq = (const float*)d_in[1];
  const float* Wk = (const float*)d_in[2];
  const float* Wv = (const float*)d_in[3];
  float* out = (float*)d_out;

  const size_t qkv_elems = (size_t)BB * NHEAD * TT * HSIZE;  // 8M halves each
  _Float16* qh  = (_Float16*)d_ws;
  _Float16* kh  = qh + qkv_elems;
  _Float16* vth = kh + qkv_elems;
  _Float16* xh  = vth + qkv_elems;                           // [B][T][1024]
  _Float16* wh  = xh + (size_t)BB * TT * EMBED;              // [3][16][64][1024]

  cvt_x_kernel<<<(BB * TT * EMBED / 4 + 255) / 256, 256, 0, stream>>>(x, xh);
  cvt_w_kernel<<<(3 * NHEAD * HSIZE * EMBED / 4 + 255) / 256, 256, 0, stream>>>(
      Wq, Wk, Wv, wh);

  // 12288 waves / 4 per block
  qkv_proj_kernel<<<3072, 128, 0, stream>>>(xh, wh, qh, kh, vth);
  // B*H*(T/64) = 2048 blocks
  attn_kernel<<<2048, 128, 0, stream>>>(qh, kh, vth, out);

  const size_t padN = (size_t)BB * TT * (MAXE - EMBED);
  pad_zero_kernel<<<(int)((padN + 255) / 256), 256, 0, stream>>>(out);
}
